// SharedMoE_63702954934786
// MI455X (gfx1250) — matrix operand order repaired and checked by reading, not compiled, not run on hardware
//
#include <hip/hip_runtime.h>
#include <hip/hip_bf16.h>
#include <math.h>

// ---------------------------------------------------------------------------
// Shapes
// ---------------------------------------------------------------------------
constexpr int D    = 1024;          // model dim
constexpr int H    = 4096;          // hidden dim
constexpr int E    = 8;             // experts
constexpr int NTOK = 4096;          // 4 * 1024 tokens
constexpr int LOGITS_OFF = NTOK * D;            // f32 elements into d_out
constexpr int PROBS_OFF  = LOGITS_OFF + NTOK * E;

// Workspace layout (bytes). Total 136 MB of scratch assumed available.
constexpr size_t XBF_OFF   = 0;                         // NTOK*D bf16  (8 MB)
constexpr size_t W1T_OFF   = (size_t)NTOK * D * 2;      //              (8 MB)
constexpr size_t W1T_BYTES = (size_t)E * H * D * 2;     //             (64 MB)
constexpr size_t W2T_OFF   = W1T_OFF + W1T_BYTES;       //        (at 72 MB)

typedef __attribute__((ext_vector_type(16))) __bf16 v16bf;
typedef __attribute__((ext_vector_type(8)))  float  v8f;

// ---------------------------------------------------------------------------
// Prep kernel 1: x (f32) -> xbf (bf16), vectorized 8 elements / thread.
// ---------------------------------------------------------------------------
__global__ __launch_bounds__(256) void x_to_bf16(const float* __restrict__ x,
                                                 __bf16* __restrict__ xbf) {
  const size_t i8 = ((size_t)blockIdx.x * 256 + threadIdx.x) * 8;
  const float4 a = *(const float4*)(x + i8);
  const float4 b = *(const float4*)(x + i8 + 4);
  union { __bf16 h[8]; uint4 q; } u;
  u.h[0] = (__bf16)a.x; u.h[1] = (__bf16)a.y;
  u.h[2] = (__bf16)a.z; u.h[3] = (__bf16)a.w;
  u.h[4] = (__bf16)b.x; u.h[5] = (__bf16)b.y;
  u.h[6] = (__bf16)b.z; u.h[7] = (__bf16)b.w;
  *(uint4*)(xbf + i8) = u.q;
}

// ---------------------------------------------------------------------------
// Prep kernel 2: LDS-tiled transpose+convert of the expert weights.
//   W1[e][d][h] (f32) -> W1T[e][h][d] (bf16)   (B-operand layout for GEMM1)
//   W2[e][h][d] (f32) -> W2T[e][d][h] (bf16)   (B-operand layout for GEMM2)
// ---------------------------------------------------------------------------
__global__ __launch_bounds__(256) void w_transpose(
    const float* __restrict__ W1, const float* __restrict__ W2,
    __bf16* __restrict__ W1T, __bf16* __restrict__ W2T) {
  __shared__ float ft[64][68];
  const int t = threadIdx.x;
  int bid = blockIdx.x;

  const float* src; __bf16* dst; int R, C, r0, c0;
  if (bid < 8192) {                 // W1: R = D rows, C = H cols
    const int e = bid >> 10, rem = bid & 1023;
    R = D; C = H;
    r0 = (rem & 15) * 64; c0 = (rem >> 4) * 64;
    src = W1 + (size_t)e * D * H;
    dst = W1T + (size_t)e * H * D;
  } else {                          // W2: R = H rows, C = D cols
    bid -= 8192;
    const int e = bid >> 10, rem = bid & 1023;
    R = H; C = D;
    r0 = (rem >> 4) * 64; c0 = (rem & 15) * 64;
    src = W2 + (size_t)e * H * D;
    dst = W2T + (size_t)e * D * H;
  }

  const int rr = t >> 4, c4 = (t & 15) * 4;
#pragma unroll
  for (int p = 0; p < 4; ++p) {
    const int row = rr + p * 16;
    const float4 v = *(const float4*)(src + (size_t)(r0 + row) * C + c0 + c4);
    ft[row][c4 + 0] = v.x; ft[row][c4 + 1] = v.y;
    ft[row][c4 + 2] = v.z; ft[row][c4 + 3] = v.w;
  }
  __syncthreads();

  const int cc = t >> 2, r16 = (t & 3) * 16;
  union { __bf16 h[16]; uint4 q[2]; } u;
#pragma unroll
  for (int j = 0; j < 16; ++j) u.h[j] = (__bf16)ft[r16 + j][cc];
  uint4* o = (uint4*)(dst + (size_t)(c0 + cc) * R + r0 + r16);
  o[0] = u.q[0];
  o[1] = u.q[1];
}

// ---------------------------------------------------------------------------
// Router: logits = x @ Wr + br ; probs = softmax(logits). One wave per token.
// ---------------------------------------------------------------------------
__global__ __launch_bounds__(128) void moe_router(
    const float* __restrict__ x, const float* __restrict__ Wr,
    const float* __restrict__ br, float* __restrict__ out) {
  const int wave = threadIdx.x >> 5;
  const int lane = threadIdx.x & 31;
  const int tok  = blockIdx.x * 4 + wave;

  float acc[E];
#pragma unroll
  for (int e = 0; e < E; ++e) acc[e] = 0.f;

  for (int i = 0; i < D / 32; ++i) {
    const int d = lane + 32 * i;
    const float xv = x[(size_t)tok * D + d];
#pragma unroll
    for (int e = 0; e < E; ++e) acc[e] += xv * Wr[d * E + e];
  }
#pragma unroll
  for (int e = 0; e < E; ++e) {
#pragma unroll
    for (int off = 16; off > 0; off >>= 1)
      acc[e] += __shfl_down(acc[e], off, 32);
  }
  if (lane == 0) {
    float lg[E], mx = -3.4e38f;
#pragma unroll
    for (int e = 0; e < E; ++e) { lg[e] = acc[e] + br[e]; mx = fmaxf(mx, lg[e]); }
    float p[E], s = 0.f;
#pragma unroll
    for (int e = 0; e < E; ++e) { p[e] = expf(lg[e] - mx); s += p[e]; }
    const float inv = 1.f / s;
#pragma unroll
    for (int e = 0; e < E; ++e) {
      out[LOGITS_OFF + tok * E + e] = lg[e];
      out[PROBS_OFF  + tok * E + e] = p[e] * inv;
    }
  }
}

// ---------------------------------------------------------------------------
// Fragment loaders. Row stride 72 bf16 = 144 B (16-B aligned rows).
// A 16x32 bf16: lane m = L%16; half h = L/16 holds K = h*8..+7 and +16.
// B 32x16 bf16: lane n = L%16; half h = L/16 holds K = h*8..+7 and +16.
// ---------------------------------------------------------------------------
#define ROW_STRIDE 72

__device__ __forceinline__ v16bf ld_a_frag(const __bf16* base, int row0,
                                           int kofs, int lane) {
  const __bf16* p = base + (size_t)(row0 + (lane & 15)) * ROW_STRIDE + kofs +
                    ((lane >> 4) * 8);
  union { v16bf v; uint4 q[2]; } u;
  u.q[0] = *reinterpret_cast<const uint4*>(p);
  u.q[1] = *reinterpret_cast<const uint4*>(p + 16);
  return u.v;
}

__device__ __forceinline__ v16bf ld_b_frag(const __bf16* base, int row0,
                                           int kofs, int lane) {
  const __bf16* p = base + (size_t)(row0 + (lane & 15)) * ROW_STRIDE + kofs +
                    ((lane >> 4) * 8);
  union { v16bf v; uint4 q[2]; } u;
  u.q[0] = *reinterpret_cast<const uint4*>(p);
  u.q[1] = *reinterpret_cast<const uint4*>(p + 16);
  return u.v;
}

// A-fragment straight from global (row-major bf16, K contiguous).
__device__ __forceinline__ v16bf ld_a_frag_g(const __bf16* p) {
  union { v16bf v; uint4 q[2]; } u;
  u.q[0] = *reinterpret_cast<const uint4*>(p);
  u.q[1] = *reinterpret_cast<const uint4*>(p + 16);
  return u.v;
}

__device__ __forceinline__ float gelu_exact(float v) {
  return 0.5f * v * (1.f + erff(v * 0.70710678118654752f));
}

// Async global->LDS 16-byte copy (ASYNCcnt-tracked DMA path).
__device__ __forceinline__ void async_copy_b128(uint32_t lds_off, uint32_t goff,
                                                const void* sbase) {
  asm volatile("global_load_async_to_lds_b128 %0, %1, %2 offset:0"
               :: "v"(lds_off), "v"(goff),
                  "s"((unsigned long long)(uintptr_t)sbase)
               : "memory");
}
__device__ __forceinline__ void async_wait0() {
  asm volatile("s_wait_asynccnt 0x0" ::: "memory");
}
__device__ __forceinline__ void async_wait1() {
  asm volatile("s_wait_asynccnt 0x1" ::: "memory");
}
__device__ __forceinline__ void async_wait2() {
  asm volatile("s_wait_asynccnt 0x2" ::: "memory");
}

// ---------------------------------------------------------------------------
// Fused dense-MoE kernel. Block = 512 threads = 16 waves, 64-token tile.
// Double-buffered async staging: tile i+1 DMA overlaps tile i WMMA.
// ---------------------------------------------------------------------------
__global__ __launch_bounds__(512) void moe_experts(
    const __bf16* __restrict__ xbf, const __bf16* __restrict__ W1T,
    const float* __restrict__ b1,   const __bf16* __restrict__ W2T,
    const float* __restrict__ b2,   float* __restrict__ out) {
  __shared__ __align__(16) __bf16 ws1[2][64][ROW_STRIDE];   // W1^T [n][k]
  __shared__ __align__(16) __bf16 ws2[2][128][ROW_STRIDE];  // W2^T [n][k]
  __shared__ __align__(16) __bf16 hs[64][ROW_STRIDE];       // h'   [m][kh]

  const int tid     = threadIdx.x;
  const int w       = tid >> 5;      // wave 0..15
  const int lane    = tid & 31;
  const int n_local = lane & 15;
  const int half    = lane >> 4;
  const int mt      = w & 3;         // wave's M-tile (GEMM1 & GEMM2)
  const int ng      = w >> 2;        // wave's N-group 0..3
  const int tok0    = blockIdx.x * 64;

  // per-thread staging coordinates: one 16-B chunk per (row, k-octet)
  const int srow = tid >> 3;         // 0..63
  const int sk0  = (tid & 7) * 8;    // 0..56

  // per-lane base of this wave's x rows (A operand of GEMM1), bf16 global
  const __bf16* xrow =
      xbf + (size_t)(tok0 + mt * 16 + n_local) * D + half * 8;

  v8f zero;
#pragma unroll
  for (int i = 0; i < 8; ++i) zero[i] = 0.f;

  v8f acc[16];
#pragma unroll
  for (int j = 0; j < 16; ++j) acc[j] = zero;

#pragma unroll 1
  for (int e = 0; e < E; ++e) {
    // router probs for this wave's 8 output rows, this expert
    float pv[8];
#pragma unroll
    for (int r = 0; r < 8; ++r)
      pv[r] = out[PROBS_OFF + (size_t)(tok0 + mt * 16 + r + 8 * half) * E + e];

#pragma unroll 1
    for (int hc = 0; hc < H / 64; ++hc) {
      const float b1v = b1[(size_t)e * H + hc * 64 + ng * 16 + n_local];

      // ---------------- GEMM1: h-tile (16x16), K = 1024 ----------------
      // prologue: stage ws1 buf0, prefetch A-frags for kc = 0
      async_copy_b128((uint32_t)(uintptr_t)&ws1[0][srow][sk0],
                      (uint32_t)((((e * H + hc * 64 + srow) * D) + sk0) * 2),
                      W1T);
      v16bf a_cur0 = ld_a_frag_g(xrow);
      v16bf a_cur1 = ld_a_frag_g(xrow + 32);

      v8f hacc = zero;
#pragma unroll 1
      for (int kc = 0; kc < D / 64; ++kc) {
        __syncthreads();                       // other buffer free
        v16bf a_nxt0, a_nxt1;
        if (kc < 15) {
          async_copy_b128(
              (uint32_t)(uintptr_t)&ws1[(kc + 1) & 1][srow][sk0],
              (uint32_t)((((e * H + hc * 64 + srow) * D) +
                          (kc + 1) * 64 + sk0) * 2),
              W1T);
          a_nxt0 = ld_a_frag_g(xrow + (kc + 1) * 64);
          a_nxt1 = ld_a_frag_g(xrow + (kc + 1) * 64 + 32);
          async_wait1();                       // current buffer landed
        } else {
          async_wait0();
        }
        __syncthreads();                       // ... for all waves

        const __bf16* wbuf = &ws1[kc & 1][0][0];
        {
          const v16bf b0 = ld_b_frag(wbuf, ng * 16, 0, lane);
          hacc = __builtin_amdgcn_wmma_f32_16x16x32_bf16(
              false, a_cur0, false, b0, (short)0, hacc, false, false);
          const v16bf b1f = ld_b_frag(wbuf, ng * 16, 32, lane);
          hacc = __builtin_amdgcn_wmma_f32_16x16x32_bf16(
              false, a_cur1, false, b1f, (short)0, hacc, false, false);
        }
        a_cur0 = a_nxt0;
        a_cur1 = a_nxt1;
      }

      // stage ws2 buf0 now: DMA overlaps the GELU epilogue below
#pragma unroll
      for (int p = 0; p < 2; ++p) {
        const int nn = srow + p * 64;
        async_copy_b128((uint32_t)(uintptr_t)&ws2[0][nn][sk0],
                        (uint32_t)((((e * D + nn) * H) + hc * 64 + sk0) * 2),
                        W2T);
      }

      // bias + exact GELU + fold router prob, store bf16 h' to LDS
#pragma unroll
      for (int r = 0; r < 8; ++r) {
        const int m = mt * 16 + r + 8 * half;
        float v = gelu_exact(hacc[r] + b1v) * pv[r];
        hs[m][ng * 16 + n_local] = (__bf16)v;
      }
      __syncthreads();   // hs complete

      // ---------------- GEMM2: out += h' @ W2[chunk, :] ----------------
      v16bf a20 = ld_a_frag(&hs[0][0], mt * 16, 0, lane);
      v16bf a21 = ld_a_frag(&hs[0][0], mt * 16, 32, lane);

#pragma unroll
      for (int ns = 0; ns < 8; ++ns) {        // 128-column sub-chunks of D
        __syncthreads();                      // other ws2 buffer free
        if (ns < 7) {
#pragma unroll
          for (int p = 0; p < 2; ++p) {
            const int nn = srow + p * 64;
            async_copy_b128(
                (uint32_t)(uintptr_t)&ws2[(ns + 1) & 1][nn][sk0],
                (uint32_t)((((e * D + (ns + 1) * 128 + nn) * H) +
                            hc * 64 + sk0) * 2),
                W2T);
          }
          async_wait2();                      // current ws2 buffer landed
        } else {
          async_wait0();
        }
        __syncthreads();                      // ... for all waves

        const __bf16* wbuf = &ws2[ns & 1][0][0];
#pragma unroll
        for (int j01 = 0; j01 < 2; ++j01) {
          const int j    = ns * 2 + j01;      // compile-time (unrolled)
          const int lrow = ng * 16 + 64 * j01;
          const v16bf bb0 = ld_b_frag(wbuf, lrow, 0, lane);
          acc[j] = __builtin_amdgcn_wmma_f32_16x16x32_bf16(
              false, a20, false, bb0, (short)0, acc[j], false, false);
          const v16bf bb1 = ld_b_frag(wbuf, lrow, 32, lane);
          acc[j] = __builtin_amdgcn_wmma_f32_16x16x32_bf16(
              false, a21, false, bb1, (short)0, acc[j], false, false);
        }
      }
    }
  }

  // -------- epilogue: + sum_e probs * b2[e], store mixed output --------
#pragma unroll
  for (int j = 0; j < 16; ++j) {
    const int col = (ng + 4 * j) * 16 + n_local;
#pragma unroll
    for (int r = 0; r < 8; ++r) {
      const int m = mt * 16 + r + 8 * half;
      float bias = 0.f;
#pragma unroll
      for (int e = 0; e < E; ++e)
        bias += out[PROBS_OFF + (size_t)(tok0 + m) * E + e] *
                b2[(size_t)e * D + col];
      out[(size_t)(tok0 + m) * D + col] = acc[j][r] + bias;
    }
  }
}

// ---------------------------------------------------------------------------
// Launch
// ---------------------------------------------------------------------------
extern "C" void kernel_launch(void* const* d_in, const int* in_sizes, int n_in,
                              void* d_out, int out_size, void* d_ws,
                              size_t ws_size, hipStream_t stream) {
  (void)in_sizes; (void)n_in; (void)ws_size; (void)out_size;
  const float* x  = (const float*)d_in[0];
  const float* Wr = (const float*)d_in[1];
  const float* br = (const float*)d_in[2];
  const float* W1 = (const float*)d_in[3];
  const float* b1 = (const float*)d_in[4];
  const float* W2 = (const float*)d_in[5];
  const float* b2 = (const float*)d_in[6];
  float* out = (float*)d_out;

  __bf16* xbf = (__bf16*)((char*)d_ws + XBF_OFF);
  __bf16* W1T = (__bf16*)((char*)d_ws + W1T_OFF);
  __bf16* W2T = (__bf16*)((char*)d_ws + W2T_OFF);

  x_to_bf16 <<<(NTOK * D) / (256 * 8), 256, 0, stream>>>(x, xbf);
  w_transpose<<<16384, 256, 0, stream>>>(W1, W2, W1T, W2T);
  moe_router <<<NTOK / 4, 128, 0, stream>>>(x, Wr, br, out);
  moe_experts<<<NTOK / 64, 512, 0, stream>>>(xbf, W1T, b1, W2T, b2, out);
}
